// CRFModel_22462678958243
// MI455X (gfx1250) — matrix operand order repaired and checked by reading, not compiled, not run on hardware
//
#include <hip/hip_runtime.h>

#define B_SZ 512
#define T_SZ 512
#define K_SZ 64
#define LOG2E 1.4426950408889634f
#define LN2   0.6931471805599453f

typedef __attribute__((ext_vector_type(16))) _Float16 v16h;
typedef __attribute__((ext_vector_type(8)))  float    v8f;

// ---------------------------------------------------------------------------
// Forward (partition function) kernel.
// One workgroup = 128 threads = 4 wave32 owns 16 batch rows for all T steps.
// alpha'[b,j] = em[b,t,j] + m_b + ln2 * log2( P(b,:) * E(:,j) )
//   P = exp2((alpha - rowmax) * log2e) in f16, E = exp(trans) in f16,
//   f32 accumulation via v_wmma_f32_16x16x32_f16 (2 k-slices for K=64).
// Raw v_exp_f32 / v_log_f32 are safe: acc in [~0.9, 70], exp2 arg <= 0.
// ---------------------------------------------------------------------------
__global__ __launch_bounds__(128) void crf_forward_kernel(
    const float* __restrict__ em, const unsigned char* __restrict__ mask,
    const float* __restrict__ start_tr, const float* __restrict__ end_tr,
    const float* __restrict__ trans, float* __restrict__ denom)
{
  __shared__ float    s_alpha[16][K_SZ + 4];   // stride 272B: 16B aligned, conflict-free
  __shared__ _Float16 s_p[16][K_SZ + 2];       // P in f16, stride 132B
  __shared__ _Float16 s_E[K_SZ][K_SZ];         // exp(transitions) in f16
  __shared__ float    s_rowmax[16];
  __shared__ float    s_wmax[4][16];           // per-wave row-max partials

  const int tid  = threadIdx.x;
  const int lane = tid & 31;
  const int w    = tid >> 5;          // wave id -> owns cols 16w..16w+15
  const int n    = lane & 15;         // C/D layout: N = lane mod 16
  const int g    = lane >> 4;         // lane half (selects K/M range)
  const int col  = w * 16 + n;
  const int r0   = blockIdx.x * 16;
  const int mrow = lane & 15;         // A layout: M = lane mod 16

  // Per-lane row base pointers in D layout (row m = v + 8g, column = col).
  const float*         emp[8];
  const unsigned char* mkp[8];
  #pragma unroll
  for (int v = 0; v < 8; ++v) {
    int m = v + 8 * g;
    emp[v] = em + ((size_t)(r0 + m) * T_SZ) * K_SZ + col;
    mkp[v] = mask + (size_t)(r0 + m) * T_SZ;
  }

  // E = exp(trans), once per block.
  for (int idx = tid; idx < K_SZ * K_SZ; idx += 128)
    s_E[idx >> 6][idx & 63] = (_Float16)__expf(trans[idx]);

  // alpha0 = start + emissions[:, 0]
  for (int idx = tid; idx < 16 * K_SZ; idx += 128) {
    int r = idx >> 6, j = idx & 63;
    s_alpha[r][j] = start_tr[j] + em[((size_t)(r0 + r) * T_SZ) * K_SZ + j];
  }
  __syncthreads();

  // B operands constant over time: hoist into registers.
  // B 32x16 f16, contiguous k order (same as A): lanes 0-15 hold K 0..15, lanes 16-31 hold K 16..31.
  v16h b0, b1;
  #pragma unroll
  for (int h = 0; h < 16; ++h) {
    b0[h] = s_E[16 * g + h][col];
    b1[h] = s_E[32 + 16 * g + h][col];
  }

  // Initial rowmax (once; simple form) + P.
  if (tid < 16) {
    float mx = -3.4e38f;
    for (int j = 0; j < K_SZ; ++j) mx = fmaxf(mx, s_alpha[tid][j]);
    s_rowmax[tid] = mx;
  }
  __syncthreads();
  {
    int r = tid >> 3, c0 = (tid & 7) * 8;
    float rmax = s_rowmax[r];
    #pragma unroll
    for (int cc = 0; cc < 8; ++cc)
      s_p[r][c0 + cc] =
          (_Float16)__builtin_amdgcn_exp2f((s_alpha[r][c0 + cc] - rmax) * LOG2E);
  }
  __syncthreads();

  // Software pipeline: preload em/mask for t = 1.
  float emc[8] = {}, mc[8] = {};
  #pragma unroll
  for (int v = 0; v < 8; ++v) {
    emc[v] = emp[v][(size_t)1 * K_SZ];
    mc[v]  = (float)mkp[v][1];
  }

  for (int t = 1; t < T_SZ; ++t) {
    // Issue next step's loads first: they drain behind the WMMAs/barriers.
    float emn[8] = {}, mn[8] = {};
    if (t + 1 < T_SZ) {
      #pragma unroll
      for (int v = 0; v < 8; ++v) {
        emn[v] = emp[v][(size_t)(t + 1) * K_SZ];
        mn[v]  = (float)mkp[v][t + 1];
      }
    }
    if (t + 2 < T_SZ)
      __builtin_prefetch(emp[0] + (size_t)(t + 2) * K_SZ, 0, 0);

    // A operands from s_p (16-bit A 16x32, contiguous order matching B):
    // k = 16*g + h.
    v16h a0, a1;
    #pragma unroll
    for (int h = 0; h < 16; ++h) {
      int k = 16 * g + h;
      a0[h] = s_p[mrow][k];
      a1[h] = s_p[mrow][32 + k];
    }
    v8f c = {};
    c = __builtin_amdgcn_wmma_f32_16x16x32_f16(false, a0, false, b0,
                                               (short)0, c, false, false);
    c = __builtin_amdgcn_wmma_f32_16x16x32_f16(false, a1, false, b1,
                                               (short)0, c, false, false);

    // Branchless masked update (D layout: vgpr v -> row m = v + 8g).
    float vals[8];
    #pragma unroll
    for (int v = 0; v < 8; ++v) {
      int m = v + 8 * g;
      float aold = s_alpha[m][col];
      float nv = emc[v] + s_rowmax[m] + LN2 * __builtin_amdgcn_logf(c[v]);
      vals[v] = aold + mc[v] * (nv - aold);
    }
    __syncthreads();                           // all reads of s_p/alpha/rowmax done

    // Write new alpha (wave-exclusive columns) + wave-local row-max partials.
    #pragma unroll
    for (int v = 0; v < 8; ++v) s_alpha[v + 8 * g][col] = vals[v];
    float pm[8];
    #pragma unroll
    for (int v = 0; v < 8; ++v) {
      float x = vals[v];
      x = fmaxf(x, __shfl_xor(x, 1, 32));
      x = fmaxf(x, __shfl_xor(x, 2, 32));
      x = fmaxf(x, __shfl_xor(x, 4, 32));
      x = fmaxf(x, __shfl_xor(x, 8, 32));      // max over the 16-lane group
      pm[v] = x;
    }
    if (n == 0) {
      #pragma unroll
      for (int v = 0; v < 8; ++v) s_wmax[w][8 * g + v] = pm[v];
    }
    __syncthreads();                           // alpha + partials visible

    // Rebuild P (one row per thread, 8 columns each) + publish rowmax.
    {
      int r = tid >> 3, c0 = (tid & 7) * 8;
      float rmax = fmaxf(fmaxf(s_wmax[0][r], s_wmax[1][r]),
                         fmaxf(s_wmax[2][r], s_wmax[3][r]));
      if ((tid & 7) == 0) s_rowmax[r] = rmax;
      #pragma unroll
      for (int cc = 0; cc < 8; ++cc)
        s_p[r][c0 + cc] =
            (_Float16)__builtin_amdgcn_exp2f((s_alpha[r][c0 + cc] - rmax) * LOG2E);
    }
    __syncthreads();

    #pragma unroll
    for (int v = 0; v < 8; ++v) { emc[v] = emn[v]; mc[v] = mn[v]; }
  }

  // denom[b] = logsumexp_j(alphaT + end_transitions)
  if (tid < 16) {
    float mx = -3.4e38f;
    for (int j = 0; j < K_SZ; ++j) mx = fmaxf(mx, s_alpha[tid][j] + end_tr[j]);
    float s = 0.f;
    for (int j = 0; j < K_SZ; ++j)
      s += __builtin_amdgcn_exp2f((s_alpha[tid][j] + end_tr[j] - mx) * LOG2E);
    denom[r0 + tid] = mx + LN2 * __builtin_amdgcn_logf(s);
  }
}

// ---------------------------------------------------------------------------
// Numerator score: one thread per batch element, branchless mask.
// ---------------------------------------------------------------------------
__global__ void crf_score_kernel(
    const float* __restrict__ em, const int* __restrict__ tags,
    const unsigned char* __restrict__ mask, const float* __restrict__ start_tr,
    const float* __restrict__ end_tr, const float* __restrict__ trans,
    float* __restrict__ score)
{
  int b = blockIdx.x * blockDim.x + threadIdx.x;
  if (b >= B_SZ) return;
  const int* tg           = tags + (size_t)b * T_SZ;
  const unsigned char* mk = mask + (size_t)b * T_SZ;
  const float* eb         = em + (size_t)b * T_SZ * K_SZ;

  int prev = tg[0];
  float s  = start_tr[prev] + eb[prev];
  int cnt  = (int)mk[0];
  for (int t = 1; t < T_SZ; ++t) {
    int cur  = tg[t];
    float mf = (float)mk[t];
    s += mf * (trans[prev * K_SZ + cur] + eb[(size_t)t * K_SZ + cur]);
    cnt += (int)mk[t];
    prev = cur;
  }
  s += end_tr[tg[cnt - 1]];   // seq_ends = sum(mask) - 1
  score[b] = s;
}

// ---------------------------------------------------------------------------
// Final reduction: out = mean(denom - score) = -mean(llh)
// ---------------------------------------------------------------------------
__global__ void crf_reduce_kernel(const float* __restrict__ denom,
                                  const float* __restrict__ score,
                                  float* __restrict__ out)
{
  __shared__ float sm[256];
  float s = 0.f;
  for (int b = threadIdx.x; b < B_SZ; b += 256) s += denom[b] - score[b];
  sm[threadIdx.x] = s;
  __syncthreads();
  for (int off = 128; off > 0; off >>= 1) {
    if (threadIdx.x < off) sm[threadIdx.x] += sm[threadIdx.x + off];
    __syncthreads();
  }
  if (threadIdx.x == 0) out[0] = sm[0] / (float)B_SZ;
}

extern "C" void kernel_launch(void* const* d_in, const int* in_sizes, int n_in,
                              void* d_out, int out_size, void* d_ws, size_t ws_size,
                              hipStream_t stream) {
  const float*         em       = (const float*)d_in[0];          // (B,T,K) f32
  const int*           tags     = (const int*)d_in[1];            // (B,T) i32
  const unsigned char* mask     = (const unsigned char*)d_in[2];  // (B,T) bool (1B)
  const float*         start_tr = (const float*)d_in[3];          // (K,)
  const float*         end_tr   = (const float*)d_in[4];          // (K,)
  const float*         trans    = (const float*)d_in[5];          // (K,K)

  float* denom = (float*)d_ws;        // [B]
  float* score = denom + B_SZ;        // [B]

  crf_forward_kernel<<<B_SZ / 16, 128, 0, stream>>>(em, mask, start_tr, end_tr,
                                                    trans, denom);
  crf_score_kernel<<<(B_SZ + 255) / 256, 256, 0, stream>>>(em, tags, mask,
                                                           start_tr, end_tr,
                                                           trans, score);
  crf_reduce_kernel<<<1, 256, 0, stream>>>(denom, score, (float*)d_out);
}
